// DynamicalGraphLearning_82016695485298
// MI455X (gfx1250) — matrix operand order repaired and checked by reading, not compiled, not run on hardware
//
#include <hip/hip_runtime.h>
#include <hip/hip_bf16.h>

#define N_NODES 16384
#define DIMD    128
#define H1F     128
#define H2F     64
#define NCLS    10
#define KNN     20
#define NEG_SLOPE 0.2f
#define CAND_CAP 24
#define WAYS    4
#define CT_PER_WAY (N_NODES / 16 / WAYS)   // 256 column tiles per way
#define BROW    136                         // halves per staged node row (256B data + 16B pad)

typedef __attribute__((ext_vector_type(16))) _Float16 v16h;
typedef __attribute__((ext_vector_type(8)))  _Float16 v8h;
typedef __attribute__((ext_vector_type(8)))  float    v8f;

__device__ __forceinline__ v16h mk16(v8h lo, v8h hi) {
  v16h r;
#pragma unroll
  for (int i = 0; i < 8; ++i) { r[i] = lo[i]; r[i + 8] = hi[i]; }
  return r;
}

// ---------------- small utility kernels ----------------

__global__ void k_f32_to_f16(const float* __restrict__ s, _Float16* __restrict__ d, int n) {
  int i = blockIdx.x * blockDim.x + threadIdx.x;
  if (i < n) d[i] = (_Float16)s[i];
}

// Wt[c*R + r] = W[r*C + c]  (f32 -> f16 transposed copy, weights are tiny)
__global__ void k_transpose_f16(const float* __restrict__ W, _Float16* __restrict__ Wt, int R, int C) {
  int i = blockIdx.x * blockDim.x + threadIdx.x;
  if (i < R * C) { int r = i / C, c = i % C; Wt[(size_t)c * R + r] = (_Float16)W[i]; }
}

__global__ void k_zero2(float* a) { if (threadIdx.x < 2) a[threadIdx.x] = 0.f; }

// ---------------- generic f16 WMMA GEMM: C[M,N] = A[M,K] * Bt[N,K]^T ----------------

__global__ void k_gemm_f16(const _Float16* __restrict__ A, const _Float16* __restrict__ Bt,
                           float* __restrict__ C, int M, int N, int K) {
  int lane = threadIdx.x & 31;
  int wid  = blockIdx.x * (blockDim.x >> 5) + (threadIdx.x >> 5);
  int tiles_n = N >> 4;
  int tm = wid / tiles_n, tn = wid % tiles_n;
  if (tm >= (M >> 4)) return;
  int r0 = tm << 4, c0 = tn << 4;
  int half = lane >> 4;
  int l15  = lane & 15;
  v8f acc = {};
  for (int kb = 0; kb < K; kb += 32) {
    const _Float16* ar = A + (size_t)(r0 + l15) * K + kb + half * 8;
    v16h af = mk16(*(const v8h*)ar, *(const v8h*)(ar + 16));
    const _Float16* br = Bt + (size_t)(c0 + l15) * K + kb + half * 8;
    v16h bf = mk16(*(const v8h*)br, *(const v8h*)(br + 16));
    acc = __builtin_amdgcn_wmma_f32_16x16x32_f16(false, af, false, bf, (short)0, acc, false, false);
  }
#pragma unroll
  for (int v = 0; v < 8; ++v) {
    int row = r0 + v + half * 8;
    C[(size_t)row * N + c0 + l15] = acc[v];
  }
}

// ---------------- fused kNN partials: sim = xn @ xn^T, streaming top-20 ----------------
// Block = 8 waves sharing async-staged (double-buffered) 16-column B tiles in LDS.
// Grid is split 4 ways over the column space; partial top-20 lists merged by k_knn_merge.

__global__ void k_knn_part(const _Float16* __restrict__ xn,
                           float* __restrict__ pv, int* __restrict__ pi) {
  __shared__ _Float16 Bb[2][16 * BROW];          // staged B tiles (padded rows)
  __shared__ float topv[8][16][KNN];
  __shared__ int   topi[8][16][KNN];
  __shared__ float cndv[8][16][CAND_CAP];
  __shared__ int   cndi[8][16][CAND_CAP];
  __shared__ int   cndn[8][16];
  __shared__ float thr [8][16];

  int lane = threadIdx.x & 31;
  int w    = threadIdx.x >> 5;
  int rowgrp = blockIdx.x >> 2;                  // 128 groups of 8 row-stripes
  int way    = blockIdx.x & 3;                   // column-range split
  int tile = rowgrp * 8 + w;
  int r0   = tile << 4;
  int half = lane >> 4, l15 = lane & 15;
  int tbase = way * CT_PER_WAY;
  const float NEGINF = -__builtin_inff();

  // cooperative async stage of one 16-node x 128-half B tile (each wave moves 2 node rows)
  auto issue = [&](int buf, int ct) {
    int node = (ct << 4) + w * 2 + half;
    const _Float16* gp = xn + (size_t)node * DIMD + l15 * 8;          // 16B per lane
    unsigned ldsoff = (unsigned)(unsigned long long)
        (&Bb[buf][(w * 2 + half) * BROW + l15 * 8]);
    asm volatile("global_load_async_to_lds_b128 %0, %1, off"
                 :: "v"(ldsoff), "v"(gp) : "memory");
  };

  issue(0, tbase);                                // prefetch first tile

  if (lane < 16) {
#pragma unroll
    for (int j = 0; j < KNN; ++j) { topv[w][lane][j] = NEGINF; topi[w][lane][j] = 0; }
    thr[w][lane] = NEGINF; cndn[w][lane] = 0;
  }

  // persistent A fragments for this wave's 16 rows (K=128 -> 4 fragments)
  v16h af[4];
#pragma unroll
  for (int kc = 0; kc < 4; ++kc) {
    const _Float16* ar = xn + (size_t)(r0 + l15) * DIMD + kc * 32 + half * 8;
    af[kc] = mk16(*(const v8h*)ar, *(const v8h*)(ar + 16));
  }

  float tr[8];
#pragma unroll
  for (int v = 0; v < 8; ++v) tr[v] = NEGINF;

  for (int t = 0; t < CT_PER_WAY; ++t) {
    int buf = t & 1;
    int nt = (t + 1 < CT_PER_WAY) ? t + 1 : t;    // clamped prefetch keeps wait-count uniform
    issue(buf ^ 1, tbase + nt);
    asm volatile("s_wait_asynccnt 1" ::: "memory");  // our portion of buf done
    __syncthreads();                                  // all waves' portions done

    int c0 = (tbase + t) << 4;
    v8f acc = {};
#pragma unroll
    for (int kc = 0; kc < 4; ++kc) {
      const _Float16* bp = &Bb[buf][l15 * BROW + kc * 32 + half * 8];
      v16h bf = mk16(*(const v8h*)bp, *(const v8h*)(bp + 16));
      acc = __builtin_amdgcn_wmma_f32_16x16x32_f16(false, af[kc], false, bf, (short)0, acc, false, false);
    }
    if (c0 == r0) {                                 // exclude self-similarity
#pragma unroll
      for (int v = 0; v < 8; ++v)
        if (l15 == v + half * 8) acc[v] = NEGINF;
    }
    bool any = false;
#pragma unroll
    for (int v = 0; v < 8; ++v) {
      float val = acc[v];
      if (val > tr[v]) {                            // rare after warm-up
        int m = v + half * 8;
        int p = atomicAdd(&cndn[w][m], 1);
        if (p < CAND_CAP) { cndv[w][m][p] = val; cndi[w][m][p] = c0 + l15; }
        any = true;
      }
    }
    if (__ballot(any) != 0ull) {                    // wave-uniform merge, no block barriers inside
      asm volatile("s_wait_dscnt 0" ::: "memory");
      if (lane < 16) {
        int m = lane;
        int nc = cndn[w][m]; if (nc > CAND_CAP) nc = CAND_CAP;
        for (int i = 0; i < nc; ++i) {
          float v = cndv[w][m][i];
          float mn = topv[w][m][0]; int mp = 0;
#pragma unroll
          for (int j = 1; j < KNN; ++j) { float x = topv[w][m][j]; if (x < mn) { mn = x; mp = j; } }
          if (v > mn) { topv[w][m][mp] = v; topi[w][m][mp] = cndi[w][m][i]; }
        }
        float mn = topv[w][m][0];
#pragma unroll
        for (int j = 1; j < KNN; ++j) mn = fminf(mn, topv[w][m][j]);
        thr[w][m] = mn; cndn[w][m] = 0;
      }
      asm volatile("s_wait_dscnt 0" ::: "memory");
#pragma unroll
      for (int v = 0; v < 8; ++v) tr[v] = thr[w][v + half * 8];
    }
    __syncthreads();                                // buf[b] free for refill next iteration
  }

  asm volatile("s_wait_dscnt 0" ::: "memory");
  if (lane < 16) {
#pragma unroll
    for (int j = 0; j < KNN; ++j) {
      pv[((size_t)way * N_NODES + r0 + lane) * KNN + j] = topv[w][lane][j];
      pi[((size_t)way * N_NODES + r0 + lane) * KNN + j] = topi[w][lane][j];
    }
  }
}

// merge WAYS partial top-20 lists per row -> final neighbor indices
__global__ void k_knn_merge(const float* __restrict__ pv, const int* __restrict__ pi,
                            int* __restrict__ idx) {
  __shared__ float tv[128][KNN];
  __shared__ int   ti[128][KNN];
  int t = threadIdx.x;
  int row = blockIdx.x * 128 + t;
#pragma unroll
  for (int j = 0; j < KNN; ++j) { tv[t][j] = -__builtin_inff(); ti[t][j] = 0; }
  for (int wy = 0; wy < WAYS; ++wy) {
    for (int j = 0; j < KNN; ++j) {
      size_t off = ((size_t)wy * N_NODES + row) * KNN + j;
      float v = pv[off];
      int  id = pi[off];
      float mn = tv[t][0]; int mp = 0;
#pragma unroll
      for (int q = 1; q < KNN; ++q) { float x = tv[t][q]; if (x < mn) { mn = x; mp = q; } }
      if (v > mn) { tv[t][mp] = v; ti[t][mp] = id; }
    }
  }
#pragma unroll
  for (int j = 0; j < KNN; ++j) idx[(size_t)row * KNN + j] = ti[t][j];
}

// ---------------- epilogues ----------------

__global__ void k_gate_epi(const float* __restrict__ Cg, const float* __restrict__ gate_b,
                           const float* __restrict__ x, float* __restrict__ xg,
                           _Float16* __restrict__ xg_h, float* __restrict__ attsum) {
  int stride = gridDim.x * blockDim.x;
  float lsum = 0.f;
  for (int i = blockIdx.x * blockDim.x + threadIdx.x; i < N_NODES * DIMD; i += stride) {
    float a = 1.f / (1.f + expf(-(Cg[i] + gate_b[i & (DIMD - 1)])));
    float g = a * x[i];
    xg[i] = g; xg_h[i] = (_Float16)g;
    lsum += fabsf(a);
  }
  atomicAdd(attsum, lsum);
}

__global__ void k_norm(const float* __restrict__ xg, _Float16* __restrict__ xn) {
  int lane = threadIdx.x & 31;
  int row  = blockIdx.x * (blockDim.x >> 5) + (threadIdx.x >> 5);
  const float* r = xg + (size_t)row * DIMD;
  float ss = 0.f;
#pragma unroll
  for (int j = 0; j < 4; ++j) { float v = r[lane + 32 * j]; ss += v * v; }
#pragma unroll
  for (int off = 16; off; off >>= 1) ss += __shfl_xor(ss, off, 32);
  float inv = 1.f / (sqrtf(ss) + 1e-12f);
#pragma unroll
  for (int j = 0; j < 4; ++j) xn[(size_t)row * DIMD + lane + 32 * j] = (_Float16)(r[lane + 32 * j] * inv);
}

__global__ void k_edot(const float* __restrict__ hW, const float* __restrict__ a_src,
                       const float* __restrict__ a_dst, float* __restrict__ es,
                       float* __restrict__ ed, int F) {
  int lane = threadIdx.x & 31;
  int n = blockIdx.x * (blockDim.x >> 5) + (threadIdx.x >> 5);
  const float* r = hW + (size_t)n * F;
  float s = 0.f, d = 0.f;
  for (int j = lane; j < F; j += 32) { float v = r[j]; s += v * a_src[j]; d += v * a_dst[j]; }
#pragma unroll
  for (int off = 16; off; off >>= 1) { s += __shfl_xor(s, off, 32); d += __shfl_xor(d, off, 32); }
  if (!lane) { es[n] = s; ed[n] = d; }
}

template <int F, bool ELU>
__global__ void k_gat_agg(const float* __restrict__ hW, const float* __restrict__ es,
                          const float* __restrict__ ed, const int* __restrict__ idx,
                          float* __restrict__ out_f32, _Float16* __restrict__ out_f16) {
  int n = blockIdx.x, f = threadIdx.x;
  float e0 = ed[n];
  int nb[KNN]; float lg[KNN];
  float mx = -__builtin_inff();
#pragma unroll
  for (int j = 0; j < KNN; ++j) {
    nb[j] = idx[n * KNN + j];
    float z = e0 + es[nb[j]];
    z = z > 0.f ? z : NEG_SLOPE * z;
    lg[j] = z; mx = fmaxf(mx, z);
  }
  float s = 0.f;
#pragma unroll
  for (int j = 0; j < KNN; ++j) { lg[j] = expf(lg[j] - mx); s += lg[j]; }
  float inv = 1.f / s, acc = 0.f;
#pragma unroll
  for (int j = 0; j < KNN; ++j) acc += lg[j] * hW[(size_t)nb[j] * F + f];
  acc *= inv;
  if (ELU) {
    acc = acc > 0.f ? acc : expf(acc) - 1.f;
    out_f16[(size_t)n * F + f] = (_Float16)acc;
  } else {
    out_f32[(size_t)n * F + f] = acc;
  }
}

__global__ void k_cls(const float* __restrict__ rep, const float* __restrict__ cw,
                      const float* __restrict__ cb, const int* __restrict__ y,
                      float* __restrict__ losssum) {
  int lane = threadIdx.x & 31;
  int n = blockIdx.x * (blockDim.x >> 5) + (threadIdx.x >> 5);
  float pre = 0.f;
  if (lane < NCLS) {
    pre = cb[lane];
    for (int f = 0; f < H2F; ++f) pre += rep[(size_t)n * H2F + f] * cw[f * NCLS + lane];
  }
  float m = (lane < NCLS) ? pre : -__builtin_inff();
#pragma unroll
  for (int off = 16; off; off >>= 1) m = fmaxf(m, __shfl_xor(m, off, 32));
  float e = (lane < NCLS) ? expf(pre - m) : 0.f;
#pragma unroll
  for (int off = 16; off; off >>= 1) e += __shfl_xor(e, off, 32);
  float lse = m + logf(e);
  float py = __shfl(pre, y[n], 32);
  if (!lane) atomicAdd(losssum, lse - py);
}

__global__ void k_fin(const float* __restrict__ scal, float* __restrict__ out_loss) {
  out_loss[0] = scal[1] / (float)N_NODES + 1e-6f * (scal[0] / (float)N_NODES);
}

// ---------------- launcher ----------------

extern "C" void kernel_launch(void* const* d_in, const int* in_sizes, int n_in,
                              void* d_out, int out_size, void* d_ws, size_t ws_size,
                              hipStream_t stream) {
  (void)in_sizes; (void)n_in; (void)out_size; (void)ws_size;
  const float* x   = (const float*)d_in[0];
  const int*   y   = (const int*)d_in[1];
  const float* gw  = (const float*)d_in[2];
  const float* gb  = (const float*)d_in[3];
  const float* W1  = (const float*)d_in[4];
  const float* a1s = (const float*)d_in[5];
  const float* a1d = (const float*)d_in[6];
  const float* W2  = (const float*)d_in[7];
  const float* a2s = (const float*)d_in[8];
  const float* a2d = (const float*)d_in[9];
  const float* cw  = (const float*)d_in[10];
  const float* cb  = (const float*)d_in[11];

  char* base = (char*)d_ws;
  size_t o = 0;
  auto alloc = [&](size_t bytes) -> char* {
    char* p = base + o;
    o = (o + bytes + 255) & ~(size_t)255;
    return p;
  };
  _Float16* x_h  = (_Float16*)alloc((size_t)N_NODES * DIMD * 2);
  float*    xg   = (float*)   alloc((size_t)N_NODES * DIMD * 4);
  _Float16* xg_h = (_Float16*)alloc((size_t)N_NODES * DIMD * 2);
  _Float16* xn_h = (_Float16*)alloc((size_t)N_NODES * DIMD * 2);
  _Float16* gwt  = (_Float16*)alloc((size_t)DIMD * DIMD * 2);
  _Float16* W1t  = (_Float16*)alloc((size_t)DIMD * H1F * 2);
  _Float16* W2t  = (_Float16*)alloc((size_t)H1F * H2F * 2);
  float*    Cg   = (float*)   alloc((size_t)N_NODES * DIMD * 4);
  float*    hW1  = (float*)   alloc((size_t)N_NODES * H1F * 4);
  _Float16* h1_h = (_Float16*)alloc((size_t)N_NODES * H1F * 2);
  float*    hW2  = (float*)   alloc((size_t)N_NODES * H2F * 4);
  float*    e1s  = (float*)alloc((size_t)N_NODES * 4);
  float*    e1d  = (float*)alloc((size_t)N_NODES * 4);
  float*    e2s  = (float*)alloc((size_t)N_NODES * 4);
  float*    e2d  = (float*)alloc((size_t)N_NODES * 4);
  float*    pv   = (float*)alloc((size_t)WAYS * N_NODES * KNN * 4);
  int*      pi   = (int*)  alloc((size_t)WAYS * N_NODES * KNN * 4);
  int*      idx  = (int*)  alloc((size_t)N_NODES * KNN * 4);
  float*    scal = (float*)alloc(256);  // [0]=attsum, [1]=losssum

  float* rep      = (float*)d_out;                 // [N, H2]
  float* loss_out = rep + (size_t)N_NODES * H2F;   // scalar

  // 0. zero accumulators (ws is poisoned, must re-init every call)
  k_zero2<<<1, 32, 0, stream>>>(scal);

  // 1. f16 conversions / weight transposes
  k_f32_to_f16<<<(N_NODES * DIMD + 255) / 256, 256, 0, stream>>>(x, x_h, N_NODES * DIMD);
  k_transpose_f16<<<(DIMD * DIMD + 255) / 256, 256, 0, stream>>>(gw, gwt, DIMD, DIMD);
  k_transpose_f16<<<(DIMD * H1F + 255) / 256, 256, 0, stream>>>(W1, W1t, DIMD, H1F);
  k_transpose_f16<<<(H1F * H2F + 255) / 256, 256, 0, stream>>>(W2, W2t, H1F, H2F);

  // 2. gate GEMM + epilogue (sigmoid, xg, |att| sum)
  k_gemm_f16<<<(N_NODES / 16) * (DIMD / 16) / 8, 256, 0, stream>>>(x_h, gwt, Cg, N_NODES, DIMD, DIMD);
  k_gate_epi<<<1024, 256, 0, stream>>>(Cg, gb, x, xg, xg_h, &scal[0]);

  // 3. row-normalize -> f16; fused kNN (4-way column split, async-LDS staged) + merge
  k_norm<<<N_NODES / 8, 256, 0, stream>>>(xg, xn_h);
  k_knn_part<<<(N_NODES / 16 / 8) * WAYS, 256, 0, stream>>>(xn_h, pv, pi);
  k_knn_merge<<<N_NODES / 128, 128, 0, stream>>>(pv, pi, idx);

  // 4. GAT layer 1
  k_gemm_f16<<<(N_NODES / 16) * (H1F / 16) / 8, 256, 0, stream>>>(xg_h, W1t, hW1, N_NODES, H1F, DIMD);
  k_edot<<<N_NODES / 8, 256, 0, stream>>>(hW1, a1s, a1d, e1s, e1d, H1F);
  k_gat_agg<H1F, true><<<N_NODES, H1F, 0, stream>>>(hW1, e1s, e1d, idx, (float*)nullptr, h1_h);

  // 5. GAT layer 2 -> rep (written straight into d_out)
  k_gemm_f16<<<(N_NODES / 16) * (H2F / 16) / 8, 256, 0, stream>>>(h1_h, W2t, hW2, N_NODES, H2F, DIMD);
  k_edot<<<N_NODES / 8, 256, 0, stream>>>(hW2, a2s, a2d, e2s, e2d, H2F);
  k_gat_agg<H2F, false><<<N_NODES, H2F, 0, stream>>>(hW2, e2s, e2d, idx, rep, (_Float16*)nullptr);

  // 6. classifier + NLL loss, then finalize scalar loss
  k_cls<<<N_NODES / 8, 256, 0, stream>>>(rep, cw, cb, y, &scal[1]);
  k_fin<<<1, 1, 0, stream>>>(scal, loss_out);
}
